// ManifoldConstrainedHyperConnections_43276090474635
// MI455X (gfx1250) — matrix operand order repaired and checked by reading, not compiled, not run on hardware
//
#include <hip/hip_runtime.h>
#include <hip/hip_bf16.h>

// ---------------- problem constants ----------------
#define B_SZ   4
#define NHC    4
#define T_SZ   4096
#define D_SZ   1024
#define KPROJ  (NHC * D_SZ)     // 4096 : RMSNorm / projection reduction dim
#define EPS_RMS  1e-8f
#define EPS_SINK 1e-8f
#define N_SINK   20

// ---------------- types ----------------
typedef unsigned short ushort_t;
typedef __attribute__((ext_vector_type(16))) __bf16         v16bf;
typedef __attribute__((ext_vector_type(8)))  float          v8f;
typedef __attribute__((ext_vector_type(4))) unsigned int    v4u;
typedef __attribute__((ext_vector_type(4))) unsigned short  v4us;

union Frag16 { v16bf v; v4u q[2]; };   // 32B WMMA A/B fragment

__device__ __forceinline__ ushort_t f2bf(float f) {
  unsigned int u = __float_as_uint(f);
  u += 0x7FFFu + ((u >> 16) & 1u);     // round-to-nearest-even
  return (ushort_t)(u >> 16);
}

// ---- CDNA5 async global->LDS copy (ASYNCcnt-tracked, writes LDS directly) ----
__device__ __forceinline__ void async_b128(unsigned lds_off, const void* gaddr) {
  asm volatile("global_load_async_to_lds_b128 %0, %1, off"
               :: "v"(lds_off), "v"(gaddr) : "memory");
}
__device__ __forceinline__ void wait_async_le8() {
  asm volatile("s_wait_asynccnt 0x8" ::: "memory");
}
__device__ __forceinline__ void wait_async_0() {
  asm volatile("s_wait_asynccnt 0x0" ::: "memory");
}

// ================= kernel 0: W_f fp32 -> bf16, chunk-major transpose =================
// W2 layout: [kc = k/32][n (0..1023)][kk = k%32]  (each K-chunk slab is contiguous)
__global__ void __launch_bounds__(256)
wf_transpose_kernel(const float* __restrict__ Wf, ushort_t* __restrict__ W2) {
  const int idx = blockIdx.x * 256 + threadIdx.x;   // 0..262143
  const int k   = idx >> 8;
  const int n4  = (idx & 255) * 4;
  const float4 v = *(const float4*)(Wf + (size_t)k * D_SZ + n4);
  const int kc = k >> 5, kk = k & 31;
  ushort_t* base = W2 + ((size_t)kc * D_SZ) * 32 + kk;
  base[(size_t)(n4 + 0) * 32] = f2bf(v.x);
  base[(size_t)(n4 + 1) * 32] = f2bf(v.y);
  base[(size_t)(n4 + 2) * 32] = f2bf(v.z);
  base[(size_t)(n4 + 3) * 32] = f2bf(v.w);
}

// ================= kernel 1: per-(b,t) prep =================
__global__ void __launch_bounds__(256)
prep_kernel(const float* __restrict__ X,  const float* __restrict__ rms_w,
            const float* __restrict__ W_A, const float* __restrict__ b_A,
            const float* __restrict__ W_B, const float* __restrict__ b_B,
            const float* __restrict__ W_C, const float* __restrict__ b_C,
            const float* __restrict__ alpha_pre, const float* __restrict__ alpha_res,
            const float* __restrict__ alpha_post,
            ushort_t* __restrict__ Xin, float* __restrict__ Aout,
            float* __restrict__ Cout,   float* __restrict__ Pout) {
  __shared__ float xsh[KPROJ];        // raw X row (4 heads x 1024)
  __shared__ float wred[8 * 24];      // cross-wave reduction scratch
  __shared__ float sA[4];
  __shared__ float s_inv;

  const int tid  = threadIdx.x;
  const int lane = tid & 31;
  const int wid  = tid >> 5;
  const int b = blockIdx.x >> 12;            // / T_SZ
  const int t = blockIdx.x & (T_SZ - 1);

  float4 xv[NHC];
  float ss = 0.f;
#pragma unroll
  for (int n = 0; n < NHC; ++n) {
    const float4* p = (const float4*)(X + ((size_t)(b * NHC + n) * T_SZ + t) * D_SZ) + tid;
    xv[n] = *p;
    ((float4*)xsh)[n * 256 + tid] = xv[n];
    ss += xv[n].x * xv[n].x + xv[n].y * xv[n].y + xv[n].z * xv[n].z + xv[n].w * xv[n].w;
  }
#pragma unroll
  for (int s = 16; s > 0; s >>= 1) ss += __shfl_xor(ss, s, 32);
  if (lane == 0) wred[wid] = ss;
  __syncthreads();
  if (tid == 0) {
    float tot = 0.f;
    for (int w = 0; w < 8; ++w) tot += wred[w];
    s_inv = rsqrtf(tot * (1.0f / KPROJ) + EPS_RMS);
  }
  __syncthreads();
  const float inv_rms = s_inv;

  float prt[24];
#pragma unroll
  for (int c = 0; c < 24; ++c) prt[c] = 0.f;
#pragma unroll
  for (int n = 0; n < NHC; ++n) {
    const float* xf = (const float*)&xv[n];
#pragma unroll
    for (int j = 0; j < 4; ++j) {
      const int i = n * D_SZ + tid * 4 + j;
      const float xh = xf[j] * inv_rms * rms_w[i];
      const float* wa = W_A + (size_t)i * NHC;
      const float* wb = W_B + (size_t)i * (NHC * NHC);
      const float* wc = W_C + (size_t)i * NHC;
#pragma unroll
      for (int c = 0; c < 4; ++c)  prt[c]      += xh * wa[c];
#pragma unroll
      for (int c = 0; c < 16; ++c) prt[4 + c]  += xh * wb[c];
#pragma unroll
      for (int c = 0; c < 4; ++c)  prt[20 + c] += xh * wc[c];
    }
  }
#pragma unroll
  for (int c = 0; c < 24; ++c) {
    float v = prt[c];
#pragma unroll
    for (int s = 16; s > 0; s >>= 1) v += __shfl_xor(v, s, 32);
    if (lane == 0) wred[wid * 24 + c] = v;
  }
  __syncthreads();

  if (tid == 0) {
    float pr[24];
    for (int c = 0; c < 24; ++c) {
      float a = 0.f;
      for (int w = 0; w < 8; ++w) a += wred[w * 24 + c];
      pr[c] = a;
    }
    const float ap  = alpha_pre[0];
    const float ar  = alpha_res[0];
    const float apo = alpha_post[0];
    float A[4], C[4], P[16];
    for (int i = 0; i < 4; ++i)  A[i] = 1.f / (1.f + __expf(-(pr[i] + b_A[i]) * ap));
    for (int i = 0; i < 16; ++i) P[i] = __expf((pr[4 + i] + b_B[i]) * ar);
    for (int i = 0; i < 4; ++i)  C[i] = 2.f / (1.f + __expf(-(pr[20 + i] + b_C[i]) * apo));
    for (int it = 0; it < N_SINK; ++it) {
      for (int i = 0; i < 4; ++i) {
        const float r = 1.f / (P[i*4+0] + P[i*4+1] + P[i*4+2] + P[i*4+3] + EPS_SINK);
        for (int j = 0; j < 4; ++j) P[i*4+j] *= r;
      }
      for (int j = 0; j < 4; ++j) {
        const float r = 1.f / (P[j] + P[4+j] + P[8+j] + P[12+j] + EPS_SINK);
        for (int i = 0; i < 4; ++i) P[i*4+j] *= r;
      }
    }
    const size_t bt = (size_t)b * T_SZ + t;
    for (int i = 0; i < 4; ++i) { sA[i] = A[i]; Aout[bt*4+i] = A[i]; Cout[bt*4+i] = C[i]; }
    for (int i = 0; i < 16; ++i) Pout[bt*16+i] = P[i];
  }
  __syncthreads();

  const float a0 = sA[0], a1 = sA[1], a2 = sA[2], a3 = sA[3];
  const float* xr = xsh + tid * 4;
  v4us o;
  ushort_t* op = (ushort_t*)&o;
#pragma unroll
  for (int j = 0; j < 4; ++j) {
    const float s = a0 * xr[j] + a1 * xr[D_SZ + j] + a2 * xr[2*D_SZ + j] + a3 * xr[3*D_SZ + j];
    op[j] = f2bf(s);
  }
  *(v4us*)(Xin + ((size_t)b * T_SZ + t) * D_SZ + tid * 4) = o;
}

// ================= kernel 2: WMMA GEMM (TM=64, async-staged) + stream mixing =================
#define TM      64           // t-rows per workgroup (4 WMMA M-tiles) -> 4x less W L2 traffic
#define MTILES  (TM / 16)    // 4
#define KC      32           // K chunk (one WMMA K-step)
#define NCHUNK  (D_SZ / KC)  // 32
#define NWAVES  16
#define WSLAB   64           // N-columns owned per wave
#define KPADA   1040         // padded A row (bf16): 2080B rows, 16B-aligned, 2-way banks
#define KPADW   40           // padded W row (bf16) -> 80B, 16B-aligned
#define FPAD    1028         // padded F_out row (f32)
#define WBUF_BYTES    (WSLAB * KPADW * 2)          // 5120 per wave per buffer
#define SMEM_A_BYTES  (TM * KPADA * 2)             // 133120
#define SMEM_W_BYTES  (2 * NWAVES * WBUF_BYTES)    // 163840 (double buffered)
#define SMEM_PC_BYTES ((TM * 16 + TM * 4) * 4)     // 5120
#define SMEM_TOTAL    (SMEM_A_BYTES + SMEM_W_BYTES + SMEM_PC_BYTES)   // 302080 <= 320KB
// F_out tile (TM x FPAD f32 = 263168 B) aliases the dead A+W regions after the K sweep.

__device__ __forceinline__ void issue_w_chunk(const ushort_t* __restrict__ W2,
                                              int kc, int nbase,
                                              unsigned ldsWaveBase, int lane) {
  // wave-private slab: 64 rows x 32 k (4KB contiguous in W2), padded rows in LDS
  const ushort_t* gbase = W2 + ((size_t)kc * D_SZ + nbase) * KC;
#pragma unroll
  for (int i = 0; i < 8; ++i) {
    const int idx = lane + i * 32;       // 0..255 (16B units)
    const int nl  = idx >> 2;
    const int sub = idx & 3;
    async_b128(ldsWaveBase + (unsigned)(nl * (KPADW * 2) + sub * 16),
               gbase + (size_t)idx * 8);
  }
}

__global__ void __launch_bounds__(512)
gemm_mix_kernel(const float* __restrict__ X,
                const ushort_t* __restrict__ Xin,
                const ushort_t* __restrict__ W2,
                const float* __restrict__ Cws,
                const float* __restrict__ Pws,
                float* __restrict__ out) {
  extern __shared__ char smem[];
  ushort_t* ldsA = (ushort_t*)smem;                         // [TM][KPADA] bf16
  float*    ldsF = (float*)smem;                            // [TM][FPAD] f32 (aliases A+W)
  float*    ldsP = (float*)(smem + SMEM_A_BYTES + SMEM_W_BYTES);
  float*    ldsC = ldsP + TM * 16;

  const unsigned smemOff = (unsigned)(size_t)smem;          // LDS byte offset of base
  const unsigned aOff = smemOff;
  const unsigned wOff = smemOff + SMEM_A_BYTES;

  const int tid  = threadIdx.x;
  const int lane = tid & 31;
  const int wid  = tid >> 5;
  const int hi   = (lane >> 4) & 1;        // lane half (WMMA K/M split)
  const int b    = blockIdx.x >> 6;        // / (T_SZ/TM = 64)
  const int t0   = (blockIdx.x & 63) * TM;
  const size_t bt0 = (size_t)b * T_SZ + t0;

  // ---- async-stage A tile (X_in bf16, TM x 1024, padded rows) ----
  for (int it = 0; it < 16; ++it) {
    const int idx = tid + it * 512;        // 0..8191 over TM x 128 (16B chunks)
    const int m  = idx >> 7;
    const int c8 = idx & 127;
    async_b128(aOff + (unsigned)(m * (KPADA * 2) + c8 * 16),
               Xin + (bt0 + m) * D_SZ + c8 * 8);
  }
  // ---- stage per-row P (16/t) and C (4/t) ----
  for (int i = tid; i < TM * 16; i += 512) ldsP[i] = Pws[bt0 * 16 + i];
  if (tid < TM * 4) ldsC[tid] = Cws[bt0 * 4 + tid];

  // ---- per-wave double-buffered W pipeline ----
  const int nbase    = wid * WSLAB;
  const unsigned wb0 = wOff + (unsigned)((0 * NWAVES + wid) * WBUF_BYTES);
  const unsigned wb1 = wOff + (unsigned)((1 * NWAVES + wid) * WBUF_BYTES);

  issue_w_chunk(W2, 0, nbase, wb0, lane);
  wait_async_le8();         // A-tile copies (issued first, in-order) complete
  __syncthreads();          // A visible to all waves

  v8f acc[MTILES * 4];      // [mt][nb] accumulators: 128 VGPRs
#pragma unroll
  for (int i = 0; i < MTILES * 4; ++i)
#pragma unroll
    for (int r = 0; r < 8; ++r) acc[i][r] = 0.f;

  const ushort_t* arow0 = ldsA + (lane & 15) * KPADA + hi * 8;
  for (int kc = 0; kc < NCHUNK; ++kc) {
    const unsigned cur = (kc & 1) ? wb1 : wb0;
    if (kc + 1 < NCHUNK) {
      issue_w_chunk(W2, kc + 1, nbase, (kc & 1) ? wb0 : wb1, lane);
      wait_async_le8();     // current chunk resident, next may be in flight
    } else {
      wait_async_0();
    }

    // A fragments for the 4 M-tiles (16-bit A 16x32: lanes<16 K{0..7,16..23}, lanes>=16 +8)
    Frag16 af[MTILES];
#pragma unroll
    for (int mt = 0; mt < MTILES; ++mt) {
      const ushort_t* ar = arow0 + (mt * 16) * KPADA + kc * KC;
      af[mt].q[0] = *(const v4u*)(ar);
      af[mt].q[1] = *(const v4u*)(ar + 16);
    }

    const ushort_t* wbase = (const ushort_t*)(smem + (cur - smemOff));
#pragma unroll
    for (int i = 0; i < 4; ++i) {
      // B fragment (32x16: lanes<16 hold K {0..7,16..23}, lanes>=16 +8; n = lane%16)
      const int nl = i * 16 + (lane & 15);
      const ushort_t* br = wbase + nl * KPADW + hi * 8;
      Frag16 bfr;
      bfr.q[0] = *(const v4u*)(br);
      bfr.q[1] = *(const v4u*)(br + 16);
#pragma unroll
      for (int mt = 0; mt < MTILES; ++mt)   // B reused across 4 M-tiles
        acc[mt * 4 + i] = __builtin_amdgcn_wmma_f32_16x16x32_bf16(
            false, af[mt].v, false, bfr.v, (short)0, acc[mt * 4 + i], false, false);
    }
  }

  // ---- spill F_out tile to LDS (C/D layout: lane<16 -> M=r, lane>=16 -> M=8+r) ----
  __syncthreads();          // all waves done reading A/W (F aliases them)
#pragma unroll
  for (int mt = 0; mt < MTILES; ++mt)
#pragma unroll
    for (int i = 0; i < 4; ++i) {
      const int n = (wid * 4 + i) * 16 + (lane & 15);
#pragma unroll
      for (int r = 0; r < 8; ++r)
        ldsF[(mt * 16 + hi * 8 + r) * FPAD + n] = acc[mt * 4 + i][r];
    }
  __syncthreads();

  // ---- mixing: X_out[i] = sum_j P[i][j]*X[j] + C[i]*F_out ----
  for (int it = 0; it < TM * 256 / 512; ++it) {   // 32 iterations
    const int idx = tid + it * 512;               // over TM x 256 (float4 chunks)
    const int tl  = idx >> 8;
    const int c4  = idx & 255;
    const int d   = c4 * 4;
    const int t   = t0 + tl;
    float4 x[NHC];
#pragma unroll
    for (int j = 0; j < NHC; ++j)
      x[j] = *(const float4*)(X + ((size_t)(b * NHC + j) * T_SZ + t) * D_SZ + d);
    const float4 f = *(const float4*)(ldsF + tl * FPAD + d);
#pragma unroll
    for (int i = 0; i < NHC; ++i) {
      const float p0 = ldsP[tl * 16 + i * 4 + 0];
      const float p1 = ldsP[tl * 16 + i * 4 + 1];
      const float p2 = ldsP[tl * 16 + i * 4 + 2];
      const float p3 = ldsP[tl * 16 + i * 4 + 3];
      const float ci = ldsC[tl * 4 + i];
      float4 o;
      o.x = p0*x[0].x + p1*x[1].x + p2*x[2].x + p3*x[3].x + ci*f.x;
      o.y = p0*x[0].y + p1*x[1].y + p2*x[2].y + p3*x[3].y + ci*f.y;
      o.z = p0*x[0].z + p1*x[1].z + p2*x[2].z + p3*x[3].z + ci*f.z;
      o.w = p0*x[0].w + p1*x[1].w + p2*x[2].w + p3*x[3].w + ci*f.w;
      *(float4*)(out + ((size_t)(b * NHC + i) * T_SZ + t) * D_SZ + d) = o;
    }
  }
}

// ================= launch =================
extern "C" void kernel_launch(void* const* d_in, const int* in_sizes, int n_in,
                              void* d_out, int out_size, void* d_ws, size_t ws_size,
                              hipStream_t stream) {
  const float* X      = (const float*)d_in[0];
  const float* rms_w  = (const float*)d_in[1];
  const float* W_A    = (const float*)d_in[2];
  const float* b_A    = (const float*)d_in[3];
  const float* W_B    = (const float*)d_in[4];
  const float* b_B    = (const float*)d_in[5];
  const float* W_C    = (const float*)d_in[6];
  const float* b_C    = (const float*)d_in[7];
  const float* W_f    = (const float*)d_in[8];
  const float* a_pre  = (const float*)d_in[9];
  const float* a_res  = (const float*)d_in[10];
  const float* a_post = (const float*)d_in[11];
  float* out = (float*)d_out;

  char* ws = (char*)d_ws;
  ushort_t* W2  = (ushort_t*)ws;                               // 2 MB (chunk-major bf16 W^T)
  ushort_t* Xin = (ushort_t*)(ws + ((size_t)2 << 20));         // 32 MB
  float* Aout   = (float*)(ws + ((size_t)34 << 20));           // 256 KB
  float* Cout   = Aout + (size_t)B_SZ * T_SZ * 4;              // 256 KB
  float* Pout   = Cout + (size_t)B_SZ * T_SZ * 4;              // 1 MB

  wf_transpose_kernel<<<dim3((D_SZ * D_SZ) / 1024), dim3(256), 0, stream>>>(W_f, W2);
  prep_kernel<<<dim3(B_SZ * T_SZ), dim3(256), 0, stream>>>(
      X, rms_w, W_A, b_A, W_B, b_B, W_C, b_C, a_pre, a_res, a_post,
      Xin, Aout, Cout, Pout);
  gemm_mix_kernel<<<dim3(B_SZ * (T_SZ / TM)), dim3(512), SMEM_TOTAL, stream>>>(
      X, Xin, W2, Cout, Pout, out);
}
